// FSCTDecoderPyG_13237089206894
// MI455X (gfx1250) — compile-verified, matrix operand order checked by reading, not run on hardware
//
#include <hip/hip_runtime.h>

// ---------------- vector types for WMMA ----------------
typedef __attribute__((ext_vector_type(16))) __bf16 v16bf;
typedef __attribute__((ext_vector_type(8)))  __bf16 v8bf;
typedef __attribute__((ext_vector_type(8)))  float  v8f;
typedef __attribute__((ext_vector_type(4)))  float  v4f;

#define LDSS 40   // LDS row stride in bf16 elems: 32 data + 8 pad (keeps 16B alignment)

// ---------------- CDNA5 async global->LDS path (guarded) ----------------
#if defined(__has_builtin)
#if __has_builtin(__builtin_amdgcn_global_load_async_to_lds_b128)
#define HAS_ASYNC_LDS 1
#endif
#endif

#ifndef HAS_ASYNC_LDS
#define HAS_ASYNC_LDS 0
#endif

#if HAS_ASYNC_LDS
// Builtin prototype (from toolchain diagnostic):
//   void __builtin_amdgcn_global_load_async_to_lds_b128(
//       v4i addrspace(1)* gaddr, v4i addrspace(3)* lds, Imm int offset, Imm int cpol)
typedef int async_v4i __attribute__((vector_size(16)));
typedef __attribute__((address_space(1))) async_v4i* gbl_v4i_ptr;
typedef __attribute__((address_space(3))) async_v4i* lds_v4i_ptr;
#if __has_builtin(__builtin_amdgcn_s_wait_asynccnt)
#define WAIT_ASYNC(n) __builtin_amdgcn_s_wait_asynccnt(n)
#else
#define WAIT_ASYNC(n) asm volatile("s_wait_asynccnt %0" ::"i"(n) : "memory")
#endif
#else
#define WAIT_ASYNC(n) ((void)0)
#endif

// 16-byte global -> LDS copy (async when available)
__device__ __forceinline__ void cp16_g2l(const __bf16* g, __bf16* l) {
#if HAS_ASYNC_LDS
  __builtin_amdgcn_global_load_async_to_lds_b128(
      (gbl_v4i_ptr)g, (lds_v4i_ptr)l, /*offset=*/0, /*cpol=*/0);
#else
  *(v8bf*)l = *(const v8bf*)g;
#endif
}

// =====================================================================
// Weight convert + transpose:  Wt[n*K + k] = bf16(W[k*1024 + n])
// =====================================================================
__global__ void wt_convert_kernel(const float* __restrict__ W, __bf16* __restrict__ Wt, int K) {
  int i = blockIdx.x * blockDim.x + threadIdx.x;
  if (i >= K * 1024) return;
  int k = i >> 10;
  int n = i & 1023;
  Wt[(size_t)n * K + k] = (__bf16)W[i];
}

// =====================================================================
// kNN: one thread per query point, source positions staged in LDS.
// d2 = |q|^2 + |s|^2 - 2 q.s  (expansion form, matching the reference)
// =====================================================================
template<int KNN>
__global__ void knn_kernel(const float* __restrict__ pos_src, const float* __restrict__ pos_q,
                           int Ns, int Nq, int* __restrict__ idx_out, float* __restrict__ w_out) {
  __shared__ float sp[4096 * 3];
  int row = blockIdx.x * blockDim.x + threadIdx.x;   // b*Nq + q (Nq multiple of blockDim)
  int b = row / Nq;
  for (int i = threadIdx.x; i < Ns * 3; i += blockDim.x)
    sp[i] = pos_src[(size_t)b * Ns * 3 + i];
  __syncthreads();

  float qx = pos_q[(size_t)row * 3 + 0];
  float qy = pos_q[(size_t)row * 3 + 1];
  float qz = pos_q[(size_t)row * 3 + 2];
  float qq = qx * qx + qy * qy + qz * qz;

  float bd[KNN]; int bi[KNN];
#pragma unroll
  for (int j = 0; j < KNN; ++j) { bd[j] = 3.4e38f; bi[j] = 0; }

  for (int s = 0; s < Ns; ++s) {
    float sx = sp[s * 3 + 0], sy = sp[s * 3 + 1], sz = sp[s * 3 + 2];
    float ss = sx * sx + sy * sy + sz * sz;
    float d2 = qq + ss - 2.0f * (qx * sx + qy * sy + qz * sz);
    if (d2 < bd[KNN - 1]) {                  // strict < keeps earliest index on ties
      int j = KNN - 1;
      while (j > 0 && d2 < bd[j - 1]) { bd[j] = bd[j - 1]; bi[j] = bi[j - 1]; --j; }
      bd[j] = d2; bi[j] = s;
    }
  }
  float wsum = 0.f, wv[KNN];
#pragma unroll
  for (int j = 0; j < KNN; ++j) { wv[j] = 1.0f / fmaxf(bd[j], 1e-16f); wsum += wv[j]; }
#pragma unroll
  for (int j = 0; j < KNN; ++j) {
    idx_out[(size_t)row * KNN + j] = bi[j];
    w_out[(size_t)row * KNN + j] = wv[j] / wsum;
  }
}

// =====================================================================
// Pack: A[row, 0:1024]   = sum_j w_j * F[b, idx_j, :]   (bf16)
//       A[row, 1024:Kt]  = Xskip[row, :]                (bf16)
// One block per output row.
// =====================================================================
__global__ void pack_kernel(const float* __restrict__ F, const float* __restrict__ Xs,
                            const int* __restrict__ idx, const float* __restrict__ wgt,
                            int KNN, int Ns, int Nq, int Cs, __bf16* __restrict__ A) {
  int row = blockIdx.x;
  int b = row / Nq;
  int Kt = 1024 + Cs;
  int id[3]; float wv[3];
  for (int j = 0; j < KNN; ++j) {
    id[j] = idx[(size_t)row * KNN + j];
    wv[j] = wgt[(size_t)row * KNN + j];
  }
  const float* fb = F + (size_t)b * Ns * 1024;
  __bf16* arow = A + (size_t)row * Kt;
  for (int c = threadIdx.x; c < 1024; c += blockDim.x) {
    float acc = 0.f;
    for (int j = 0; j < KNN; ++j) acc += wv[j] * fb[(size_t)id[j] * 1024 + c];
    arow[c] = (__bf16)acc;
  }
  const float* xr = Xs + (size_t)row * Cs;
  for (int c = threadIdx.x; c < Cs; c += blockDim.x)
    arow[1024 + c] = (__bf16)xr[c];
}

// =====================================================================
// GEMM: Z[M,1024] = relu(A[M,K](bf16) * Wt[1024,K](bf16, n-major) + bias)
// Block tile 128x128, 8 waves of 64x32, K-step 32, v_wmma_f32_16x16x32_bf16.
// Double-buffered LDS; tiles staged with GLOBAL_LOAD_ASYNC_TO_LDS_B128
// (ASYNCcnt) when available, prefetching tile k+1 while computing tile k.
// CDNA5 16-bit A/B fragment layout: lane<16 -> K{0..7 | 16..23},
// lane>=16 -> K{8..15 | 24..31}; both halves are contiguous 16B runs.
// =====================================================================
__global__ __launch_bounds__(256) void gemm_bf16_kernel(
    const __bf16* __restrict__ A, const __bf16* __restrict__ Wt,
    const float* __restrict__ bias, float* __restrict__ Z, int M, int K) {
  __shared__ __bf16 sA[2][128 * LDSS];
  __shared__ __bf16 sB[2][128 * LDSS];

  const int tid = threadIdx.x;
  const int bn0 = blockIdx.x * 128;       // N tile
  const int bm0 = blockIdx.y * 128;       // M tile
  const int wave = tid >> 5;
  const int lane = tid & 31;
  const int wm = (wave & 1) * 64;         // wave M offset in block tile
  const int wn = (wave >> 1) * 32;        // wave N offset
  const int lrow = lane & 15;
  const int khalf = (lane >> 4) * 8;      // 0 or 8

  // per-thread tile-copy mapping: 2 chunks of 16B per buffer
  const int r0 = tid >> 2,            s0 = (tid & 3) * 8;          // chunk tid
  const int r1 = (tid + 256) >> 2,    s1 = ((tid + 256) & 3) * 8;  // chunk tid+256
  const __bf16* gA0 = A  + (size_t)(bm0 + r0) * K + s0;
  const __bf16* gA1 = A  + (size_t)(bm0 + r1) * K + s1;
  const __bf16* gB0 = Wt + (size_t)(bn0 + r0) * K + s0;
  const __bf16* gB1 = Wt + (size_t)(bn0 + r1) * K + s1;
  const int lA0 = r0 * LDSS + s0, lA1 = r1 * LDSS + s1;

  v8f acc[4][2] = {};

  const int nk = K >> 5;   // K / 32

  // prefetch tile 0 into buffer 0
  cp16_g2l(gA0, &sA[0][lA0]);
  cp16_g2l(gA1, &sA[0][lA1]);
  cp16_g2l(gB0, &sB[0][lA0]);
  cp16_g2l(gB1, &sB[0][lA1]);

  for (int kt = 0; kt < nk; ++kt) {
    const int cur = kt & 1;
    const bool more = (kt + 1) < nk;
    if (more) {                          // prefetch next tile into other buffer
      const int koff = (kt + 1) << 5;
      cp16_g2l(gA0 + koff, &sA[cur ^ 1][lA0]);
      cp16_g2l(gA1 + koff, &sA[cur ^ 1][lA1]);
      cp16_g2l(gB0 + koff, &sB[cur ^ 1][lA0]);
      cp16_g2l(gB1 + koff, &sB[cur ^ 1][lA1]);
      WAIT_ASYNC(4);                     // current tile done; prefetch in flight
    } else {
      WAIT_ASYNC(0);
    }
    __syncthreads();                     // tile kt visible to all waves

    v16bf afr[4], bfr[2];
#pragma unroll
    for (int i = 0; i < 4; ++i) {
      const __bf16* p = &sA[cur][(wm + i * 16 + lrow) * LDSS];
      union { v16bf v; v8bf h[2]; } u;
      u.h[0] = *(const v8bf*)(p + khalf);
      u.h[1] = *(const v8bf*)(p + 16 + khalf);
      afr[i] = u.v;
    }
#pragma unroll
    for (int j = 0; j < 2; ++j) {
      const __bf16* p = &sB[cur][(wn + j * 16 + lrow) * LDSS];
      union { v16bf v; v8bf h[2]; } u;
      u.h[0] = *(const v8bf*)(p + khalf);
      u.h[1] = *(const v8bf*)(p + 16 + khalf);
      bfr[j] = u.v;
    }
#pragma unroll
    for (int i = 0; i < 4; ++i)
#pragma unroll
      for (int j = 0; j < 2; ++j)
        acc[i][j] = __builtin_amdgcn_wmma_f32_16x16x32_bf16(
            false, afr[i], false, bfr[j], (short)0, acc[i][j], false, false);

    __syncthreads();                     // all waves done reading buf `cur`
  }

  // Epilogue: bias + ReLU -> f32 Z.  C/D layout: VGPR e = row e (lanes 0-15)
  // or row 8+e (lanes 16-31); col = lane%16.
  const int chi = (lane >> 4) * 8;
#pragma unroll
  for (int j = 0; j < 2; ++j) {
    int col = bn0 + wn + j * 16 + lrow;
    float bia = bias[col];
#pragma unroll
    for (int i = 0; i < 4; ++i) {
#pragma unroll
      for (int e = 0; e < 8; ++e) {
        int row = bm0 + wm + i * 16 + chi + e;
        float z = acc[i][j][e] + bia;
        z = fmaxf(z, 0.0f);
        Z[(size_t)row * 1024 + col] = z;
      }
    }
  }
}

// =====================================================================
// BatchNorm (training-mode batch stats), deterministic 64-way partials
// =====================================================================
__global__ void bn_partial_kernel(const float* __restrict__ Z, int M,
                                  float* __restrict__ ps, float* __restrict__ ps2) {
  int p = blockIdx.x;                  // 0..63
  int chunk = M >> 6;
  int c0 = threadIdx.x * 4;            // 256 threads * 4 channels = 1024
  v4f s = {0.f, 0.f, 0.f, 0.f}, s2 = {0.f, 0.f, 0.f, 0.f};
  int rend = (p + 1) * chunk;
  for (int r = p * chunk; r < rend; ++r) {
    v4f v = *(const v4f*)(Z + (size_t)r * 1024 + c0);
    s += v;
    s2 += v * v;
  }
#pragma unroll
  for (int j = 0; j < 4; ++j) {
    ps [p * 1024 + c0 + j] = s[j];
    ps2[p * 1024 + c0 + j] = s2[j];
  }
}

__global__ void bn_finalize_kernel(const float* __restrict__ ps, const float* __restrict__ ps2,
                                   const float* __restrict__ g, const float* __restrict__ t,
                                   int M, float* __restrict__ scale, float* __restrict__ shift) {
  int c = blockIdx.x * blockDim.x + threadIdx.x;   // 1024 total
  float s = 0.f, s2 = 0.f;
  for (int p = 0; p < 64; ++p) { s += ps[p * 1024 + c]; s2 += ps2[p * 1024 + c]; }
  float inv = 1.0f / (float)M;
  float mu = s * inv;
  float var = fmaxf(s2 * inv - mu * mu, 0.0f);
  float rs = rsqrtf(var + 1e-5f);
  float sc = g[c] * rs;
  scale[c] = sc;
  shift[c] = t[c] - sc * mu;
}

__global__ void bn_apply_kernel(const float* __restrict__ Z, const float* __restrict__ scale,
                                const float* __restrict__ shift, size_t total,
                                __bf16* __restrict__ obf, float* __restrict__ of) {
  size_t i = (size_t)blockIdx.x * blockDim.x + threadIdx.x;
  if (i >= total) return;
  int c = (int)(i & 1023);
  float y = Z[i] * scale[c] + shift[c];
  if (obf) obf[i] = (__bf16)y;
  if (of)  of[i]  = y;
}

// =====================================================================
// Host orchestration
// =====================================================================
namespace {

struct Ws {
  char* base;
  size_t off;
  void* take(size_t bytes) {
    void* p = base + off;
    off += (bytes + 255) & ~(size_t)255;
    return p;
  }
};

struct BnScratch {
  float *ps, *ps2, *scale, *shift;
};

static void run_mlp_block(const __bf16* A, int M, int K, const __bf16* Wt,
                          const float* bias, const float* g, const float* t,
                          float* Z, __bf16* out_bf, float* out_f,
                          const BnScratch& bs, hipStream_t stream) {
  dim3 gg(1024 / 128, M / 128);
  gemm_bf16_kernel<<<gg, 256, 0, stream>>>(A, Wt, bias, Z, M, K);
  bn_partial_kernel<<<64, 256, 0, stream>>>(Z, M, bs.ps, bs.ps2);
  bn_finalize_kernel<<<4, 256, 0, stream>>>(bs.ps, bs.ps2, g, t, M, bs.scale, bs.shift);
  size_t total = (size_t)M * 1024;
  bn_apply_kernel<<<(unsigned)((total + 255) / 256), 256, 0, stream>>>(
      Z, bs.scale, bs.shift, total, out_bf, out_f);
}

} // namespace

extern "C" void kernel_launch(void* const* d_in, const int* in_sizes, int n_in,
                              void* d_out, int out_size, void* d_ws, size_t ws_size,
                              hipStream_t stream) {
  (void)in_sizes; (void)n_in; (void)out_size; (void)ws_size;
  // ---- inputs (setup_inputs dict order) ----
  const float* p0 = (const float*)d_in[0];
  const float* x0 = (const float*)d_in[1];
  const float* p1 = (const float*)d_in[3];
  const float* x1 = (const float*)d_in[4];
  const float* p2 = (const float*)d_in[6];
  const float* x2 = (const float*)d_in[7];
  const float* p3 = (const float*)d_in[9];
  const float* x3 = (const float*)d_in[10];
  const float* w3a = (const float*)d_in[12]; const float* c3a = (const float*)d_in[13];
  const float* g3a = (const float*)d_in[14]; const float* t3a = (const float*)d_in[15];
  const float* w3b = (const float*)d_in[16]; const float* c3b = (const float*)d_in[17];
  const float* g3b = (const float*)d_in[18]; const float* t3b = (const float*)d_in[19];
  const float* w2a = (const float*)d_in[20]; const float* c2a = (const float*)d_in[21];
  const float* g2a = (const float*)d_in[22]; const float* t2a = (const float*)d_in[23];
  const float* w2b = (const float*)d_in[24]; const float* c2b = (const float*)d_in[25];
  const float* g2b = (const float*)d_in[26]; const float* t2b = (const float*)d_in[27];
  const float* w1a = (const float*)d_in[28]; const float* c1a = (const float*)d_in[29];
  const float* g1a = (const float*)d_in[30]; const float* t1a = (const float*)d_in[31];
  const float* w1b = (const float*)d_in[32]; const float* c1b = (const float*)d_in[33];
  const float* g1b = (const float*)d_in[34]; const float* t1b = (const float*)d_in[35];

  // ---- workspace carve-out ----
  Ws ws{(char*)d_ws, 0};
  __bf16* Wt3a = (__bf16*)ws.take((size_t)1536 * 1024 * 2);
  __bf16* Wt3b = (__bf16*)ws.take((size_t)1024 * 1024 * 2);
  __bf16* Wt2a = (__bf16*)ws.take((size_t)1280 * 1024 * 2);
  __bf16* Wt2b = (__bf16*)ws.take((size_t)1024 * 1024 * 2);
  __bf16* Wt1a = (__bf16*)ws.take((size_t)1088 * 1024 * 2);
  __bf16* Wt1b = (__bf16*)ws.take((size_t)1024 * 1024 * 2);
  int*    idxb = (int*)  ws.take((size_t)32768 * 3 * 4);
  float*  wb   = (float*)ws.take((size_t)32768 * 3 * 4);
  __bf16* Abuf = (__bf16*)ws.take((size_t)32768 * 1088 * 2);   // max M*K over all GEMM inputs
  float*  Zbuf = (float*) ws.take((size_t)32768 * 1024 * 4);
  float*  Hbuf = (float*) ws.take((size_t)8192  * 1024 * 4);   // interp feature staging
  BnScratch bs;
  bs.ps    = (float*)ws.take((size_t)64 * 1024 * 4);
  bs.ps2   = (float*)ws.take((size_t)64 * 1024 * 4);
  bs.scale = (float*)ws.take((size_t)1024 * 4);
  bs.shift = (float*)ws.take((size_t)1024 * 4);

  // ---- one-time weight convert + transpose to bf16 (n-major) ----
  auto cvt = [&](const float* W, __bf16* Wt, int K) {
    int n = K * 1024;
    wt_convert_kernel<<<(n + 255) / 256, 256, 0, stream>>>(W, Wt, K);
  };
  cvt(w3a, Wt3a, 1536); cvt(w3b, Wt3b, 1024);
  cvt(w2a, Wt2a, 1280); cvt(w2b, Wt2b, 1024);
  cvt(w1a, Wt1a, 1088); cvt(w1b, Wt1b, 1024);

  // ================= Stage 3: x3[2,256,1024] -> p2 queries (k=1), skip x2 =================
  knn_kernel<1><<<(2 * 1024) / 128, 128, 0, stream>>>(p3, p2, 256, 1024, idxb, wb);
  pack_kernel<<<2048, 256, 0, stream>>>(x3, x2, idxb, wb, 1, 256, 1024, 512, Abuf); // Kt=1536
  run_mlp_block(Abuf, 2048, 1536, Wt3a, c3a, g3a, t3a, Zbuf, Abuf, nullptr, bs, stream);
  run_mlp_block(Abuf, 2048, 1024, Wt3b, c3b, g3b, t3b, Zbuf, nullptr, Hbuf, bs, stream); // H3 [2,1024,1024]

  // ================= Stage 2: H3 -> p1 queries (k=3), skip x1 =================
  knn_kernel<3><<<(2 * 4096) / 128, 128, 0, stream>>>(p2, p1, 1024, 4096, idxb, wb);
  pack_kernel<<<8192, 256, 0, stream>>>(Hbuf, x1, idxb, wb, 3, 1024, 4096, 256, Abuf); // Kt=1280
  run_mlp_block(Abuf, 8192, 1280, Wt2a, c2a, g2a, t2a, Zbuf, Abuf, nullptr, bs, stream);
  run_mlp_block(Abuf, 8192, 1024, Wt2b, c2b, g2b, t2b, Zbuf, nullptr, Hbuf, bs, stream); // H2 [2,4096,1024]

  // ================= Stage 1: H2 -> p0 queries (k=3), skip x0 =================
  knn_kernel<3><<<(2 * 16384) / 128, 128, 0, stream>>>(p1, p0, 4096, 16384, idxb, wb);
  pack_kernel<<<32768, 256, 0, stream>>>(Hbuf, x0, idxb, wb, 3, 4096, 16384, 64, Abuf); // Kt=1088
  run_mlp_block(Abuf, 32768, 1088, Wt1a, c1a, g1a, t1a, Zbuf, Abuf, nullptr, bs, stream);
  run_mlp_block(Abuf, 32768, 1024, Wt1b, c1b, g1b, t1b, Zbuf, nullptr, (float*)d_out, bs, stream);
}
